// DeeperGCN_51805895524492
// MI455X (gfx1250) — matrix operand order repaired and checked by reading, not compiled, not run on hardware
//
#include <hip/hip_runtime.h>

// DeeperGCN forward for MI455X (gfx1250, wave32).
// - Edge passes: coalesced float4 gathers (L2-resident node state), native
//   float atomics (atomicMax via positive-float-as-uint; GLOBAL_ATOMIC_ADD_F32).
// - All GEMMs (encoder 128->64, 14x mlp 64->64, predictor 64->112) run on the
//   matrix engine via v_wmma_f32_16x16x32_bf16 (f32 accumulate).
// Workspace use: 5 * N*64 f32 node arrays + bf16 transposed weights (~64.2 MB).

#define N_NODES 50000
#define N_EDGES 800000
#define HDIM    64
#define NLAYERS 14
#define INDIM   128
#define OUTDIM  112
#define MSG_EPS 1e-7f
#define LN_EPS  1e-5f

typedef __attribute__((ext_vector_type(16))) __bf16 v16bf;
typedef __attribute__((ext_vector_type(8)))  __bf16 v8bf;
typedef __attribute__((ext_vector_type(8)))  float  v8f;

// ---------------------------------------------------------------------------
// Weight convert+transpose: W[K][ncols] f32 -> WT[ncols][K] bf16 (batched in y)
// ---------------------------------------------------------------------------
__global__ void k_wt(const float* __restrict__ W, __bf16* __restrict__ WT,
                     int K, int ncols) {
  const size_t per = (size_t)K * ncols;
  const float* w = W + per * blockIdx.y;
  __bf16* wt = WT + per * blockIdx.y;
  int t = blockIdx.x * 256 + threadIdx.x;
  if (t >= (int)per) return;
  int k = t / ncols, n = t - k * ncols;
  wt[(size_t)n * K + k] = (__bf16)w[t];
}

// ---------------------------------------------------------------------------
// Zero the three aggregation buffers (mmax / denom / num)
// ---------------------------------------------------------------------------
__global__ void k_zero(float* __restrict__ a, float* __restrict__ b,
                       float* __restrict__ c) {
  size_t t = (size_t)blockIdx.x * 256 + threadIdx.x;
  a[t] = 0.0f; b[t] = 0.0f; c[t] = 0.0f;
}

// ---------------------------------------------------------------------------
// LayerNorm + ReLU, one wave32 per node (2 features per lane, shfl_xor reduce)
// ---------------------------------------------------------------------------
__global__ void k_ln_relu(const float* __restrict__ h, const float* __restrict__ g,
                          const float* __restrict__ bl, float* __restrict__ h2) {
  int node = (int)((blockIdx.x * blockDim.x + threadIdx.x) >> 5);
  int lane = threadIdx.x & 31;
  float2 v = *(const float2*)(h + (size_t)node * HDIM + lane * 2);
  float s = v.x + v.y;
#pragma unroll
  for (int off = 16; off; off >>= 1) s += __shfl_xor(s, off, 32);
  float mu = s * (1.0f / 64.0f);
  float dx = v.x - mu, dy = v.y - mu;
  float q = dx * dx + dy * dy;
#pragma unroll
  for (int off = 16; off; off >>= 1) q += __shfl_xor(q, off, 32);
  float inv = rsqrtf(q * (1.0f / 64.0f) + LN_EPS);
  float2 o;
  o.x = fmaxf(dx * inv * g[lane * 2 + 0] + bl[lane * 2 + 0], 0.0f);
  o.y = fmaxf(dy * inv * g[lane * 2 + 1] + bl[lane * 2 + 1], 0.0f);
  *(float2*)(h2 + (size_t)node * HDIM + lane * 2) = o;
}

// ---------------------------------------------------------------------------
// Edge pass 1: msg = relu(h2[src]) + eps ; atomicMax(mmax[dst], msg)
// msg > 0 always, so positive-float bits compare correctly as uint; mmax
// initialized to 0 reproduces the reference's "empty segment -> 0" rule.
// 16 threads/edge, float4 per thread (256B coalesced gather per edge).
// ---------------------------------------------------------------------------
__global__ void k_edge_max(const long long* __restrict__ ei,
                           const float* __restrict__ h2,
                           unsigned int* __restrict__ mmax) {
  int gid = blockIdx.x * 256 + threadIdx.x;
  int e = gid >> 4, q = gid & 15;
  size_t s = (size_t)ei[e];
  size_t d = (size_t)ei[(size_t)N_EDGES + e];
  float4 v = *(const float4*)(h2 + s * HDIM + q * 4);
  unsigned int* p = mmax + d * HDIM + q * 4;
  atomicMax(p + 0, __float_as_uint(fmaxf(v.x, 0.0f) + MSG_EPS));
  atomicMax(p + 1, __float_as_uint(fmaxf(v.y, 0.0f) + MSG_EPS));
  atomicMax(p + 2, __float_as_uint(fmaxf(v.z, 0.0f) + MSG_EPS));
  atomicMax(p + 3, __float_as_uint(fmaxf(v.w, 0.0f) + MSG_EPS));
}

// ---------------------------------------------------------------------------
// Edge pass 2: ex = exp(msg - mmax[dst]); denom += ex; num += msg*ex
// ---------------------------------------------------------------------------
__global__ void k_edge_sum(const long long* __restrict__ ei,
                           const float* __restrict__ h2,
                           const float* __restrict__ mmax,
                           float* __restrict__ den, float* __restrict__ num) {
  int gid = blockIdx.x * 256 + threadIdx.x;
  int e = gid >> 4, q = gid & 15;
  size_t s = (size_t)ei[e];
  size_t d = (size_t)ei[(size_t)N_EDGES + e];
  float4 v  = *(const float4*)(h2 + s * HDIM + q * 4);
  size_t off = d * HDIM + q * 4;
  float4 mx = *(const float4*)(mmax + off);
  float m0 = fmaxf(v.x, 0.0f) + MSG_EPS, e0 = __expf(m0 - mx.x);
  float m1 = fmaxf(v.y, 0.0f) + MSG_EPS, e1 = __expf(m1 - mx.y);
  float m2 = fmaxf(v.z, 0.0f) + MSG_EPS, e2 = __expf(m2 - mx.z);
  float m3 = fmaxf(v.w, 0.0f) + MSG_EPS, e3 = __expf(m3 - mx.w);
  atomicAdd(den + off + 0, e0); atomicAdd(num + off + 0, m0 * e0);
  atomicAdd(den + off + 1, e1); atomicAdd(num + off + 1, m1 * e1);
  atomicAdd(den + off + 2, e2); atomicAdd(num + off + 2, m2 * e2);
  atomicAdd(den + off + 3, e3); atomicAdd(num + off + 3, m3 * e3);
}

// ---------------------------------------------------------------------------
// xin = h2 + num / max(denom, 1e-16)  (written in-place into num)
// ---------------------------------------------------------------------------
__global__ void k_agg(const float* __restrict__ h2, const float* __restrict__ den,
                      float* __restrict__ num) {
  size_t t = (size_t)blockIdx.x * 256 + threadIdx.x;
  num[t] = h2[t] + num[t] / fmaxf(den[t], 1e-16f);
}

// ---------------------------------------------------------------------------
// WMMA GEMM: out[M][ncols] = A[M][K] @ W[K][ncols] + bias (+ res)
// A: f32 row-major (converted to bf16 fragments in-register)
// WT: bf16, pre-transposed [n][k] so B fragments are two 16B vector loads
// One wave per 16-row tile; ntile outer, K inner (one live v8f accumulator).
// Fragment layouts per CDNA5 ISA 7.12.2:
//   A 16x32: lane l -> row m=l&15; K pairs {8*hi..8*hi+7, 16+8*hi..+7}, hi=l>>4
//   B 32x16: lane l -> col n=l&15; K {8*hi..8*hi+7, 16+8*hi..+7} (same as A)
//   C 16x16: vgpr r -> row r+8*hi, col l&15
// ---------------------------------------------------------------------------
__global__ void k_gemm(const float* __restrict__ A, const __bf16* __restrict__ WT,
                       const float* __restrict__ bias, const float* __restrict__ res,
                       float* __restrict__ out, int K, int ncols, int mtiles) {
  int wave = (int)((blockIdx.x * blockDim.x + threadIdx.x) >> 5);
  int lane = threadIdx.x & 31;
  if (wave >= mtiles) return;            // uniform per-wave: EXEC all-1s inside
  int m  = lane & 15;
  int hi = lane >> 4;
  size_t row0 = (size_t)wave * 16;
  const float* arow = A + (row0 + m) * (size_t)K;
  int ntiles = ncols >> 4;
  for (int nt = 0; nt < ntiles; ++nt) {
    int n0 = nt << 4;
    const __bf16* wrow = WT + (size_t)(n0 + m) * K;
    v8f c = {};
    for (int k0 = 0; k0 < K; k0 += 32) {
      // ---- A fragment: 4x float4 (16B, aligned), cvt f32->bf16 ----
      float4 u0 = *(const float4*)(arow + k0 + 8 * hi);
      float4 u1 = *(const float4*)(arow + k0 + 8 * hi + 4);
      float4 u2 = *(const float4*)(arow + k0 + 16 + 8 * hi);
      float4 u3 = *(const float4*)(arow + k0 + 16 + 8 * hi + 4);
      v16bf a;
      a[0]  = (__bf16)u0.x; a[1]  = (__bf16)u0.y; a[2]  = (__bf16)u0.z; a[3]  = (__bf16)u0.w;
      a[4]  = (__bf16)u1.x; a[5]  = (__bf16)u1.y; a[6]  = (__bf16)u1.z; a[7]  = (__bf16)u1.w;
      a[8]  = (__bf16)u2.x; a[9]  = (__bf16)u2.y; a[10] = (__bf16)u2.z; a[11] = (__bf16)u2.w;
      a[12] = (__bf16)u3.x; a[13] = (__bf16)u3.y; a[14] = (__bf16)u3.z; a[15] = (__bf16)u3.w;
      // ---- B fragment: two 16B bf16 vector loads from transposed weights ----
      v8bf b0 = *(const v8bf*)(wrow + k0 + 8 * hi);
      v8bf b1 = *(const v8bf*)(wrow + k0 + 16 + 8 * hi);
      v16bf b;
#pragma unroll
      for (int j = 0; j < 8; ++j) { b[j] = b0[j]; b[8 + j] = b1[j]; }
      c = __builtin_amdgcn_wmma_f32_16x16x32_bf16(
              /*neg_a=*/false, a, /*neg_b=*/false, b,
              /*c_mod=*/(short)0, c, /*reuse_a=*/false, /*reuse_b=*/false);
    }
    // ---- epilogue: bias + optional residual ----
#pragma unroll
    for (int r = 0; r < 8; ++r) {
      size_t idx = (row0 + r + 8 * hi) * (size_t)ncols + n0 + m;
      float v = c[r] + bias[n0 + m];
      if (res) v += res[idx];
      out[idx] = v;
    }
  }
}

// ---------------------------------------------------------------------------
extern "C" void kernel_launch(void* const* d_in, const int* in_sizes, int n_in,
                              void* d_out, int out_size, void* d_ws, size_t ws_size,
                              hipStream_t stream) {
  const float*     x      = (const float*)d_in[0];
  const long long* ei     = (const long long*)d_in[1];   // int64 edge_index [2][E]
  const float*     enc_W  = (const float*)d_in[2];
  const float*     enc_b  = (const float*)d_in[3];
  const float*     ln_g   = (const float*)d_in[4];
  const float*     ln_b   = (const float*)d_in[5];
  const float*     mlp_W  = (const float*)d_in[6];
  const float*     mlp_b  = (const float*)d_in[7];
  const float*     pred_W = (const float*)d_in[8];
  const float*     pred_b = (const float*)d_in[9];
  float* out = (float*)d_out;

  const size_t N64 = (size_t)N_NODES * HDIM;
  float* h    = (float*)d_ws;
  float* h2   = h    + N64;
  float* mmax = h2   + N64;
  float* den  = mmax + N64;
  float* num  = den  + N64;
  __bf16* encWT  = (__bf16*)(num + N64);
  __bf16* mlpWT  = encWT + (size_t)INDIM * HDIM;
  __bf16* predWT = mlpWT + (size_t)NLAYERS * HDIM * HDIM;

  // --- stage bf16 transposed weights (deterministic, rebuilt every call) ---
  k_wt<<<dim3((INDIM * HDIM + 255) / 256, 1), 256, 0, stream>>>(enc_W, encWT, INDIM, HDIM);
  k_wt<<<dim3((HDIM * HDIM + 255) / 256, NLAYERS), 256, 0, stream>>>(mlp_W, mlpWT, HDIM, HDIM);
  k_wt<<<dim3((HDIM * OUTDIM + 255) / 256, 1), 256, 0, stream>>>(pred_W, predWT, HDIM, OUTDIM);

  const int MT = N_NODES / 16;            // 3125 row tiles (exact)
  const int gemmBlocks = (MT + 7) / 8;    // 8 waves per 256-thread block
  const int edgeBlocks = N_EDGES * 16 / 256;
  const int nodeBlocks = (int)(N64 / 256);

  // --- encoder: h = x @ enc_W + enc_b ---
  k_gemm<<<gemmBlocks, 256, 0, stream>>>(x, encWT, enc_b, nullptr, h, INDIM, HDIM, MT);

  for (int l = 0; l < NLAYERS; ++l) {
    const float* h2p;
    if (l > 0) {
      k_ln_relu<<<N_NODES / 8, 256, 0, stream>>>(h, ln_g + (size_t)(l - 1) * HDIM,
                                                 ln_b + (size_t)(l - 1) * HDIM, h2);
      h2p = h2;
    } else {
      h2p = h;                             // first GENConv has no pre-norm
    }
    k_zero<<<nodeBlocks, 256, 0, stream>>>(mmax, den, num);
    k_edge_max<<<edgeBlocks, 256, 0, stream>>>(ei, h2p, (unsigned int*)mmax);
    k_edge_sum<<<edgeBlocks, 256, 0, stream>>>(ei, h2p, mmax, den, num);
    k_agg<<<nodeBlocks, 256, 0, stream>>>(h2p, den, num);   // num := h2 + agg
    // h = (h2 + agg) @ W_l + b_l (+ h residual for l>0); safe in-place: A=num
    k_gemm<<<gemmBlocks, 256, 0, stream>>>(num, mlpWT + (size_t)l * HDIM * HDIM,
                                           mlp_b + (size_t)l * HDIM,
                                           (l > 0) ? h : nullptr, h, HDIM, HDIM, MT);
  }

  // --- predictor: out = h @ pred_W + pred_b ---
  k_gemm<<<gemmBlocks, 256, 0, stream>>>(h, predWT, pred_b, nullptr, out, HDIM, OUTDIM, MT);

  (void)in_sizes; (void)n_in; (void)out_size; (void)ws_size;
}
